// GNNDualModule_89215060672586
// MI455X (gfx1250) — matrix operand order repaired and checked by reading, not compiled, not run on hardware
//
#include <hip/hip_runtime.h>
#include <float.h>

// ---------- vector typedefs ----------
typedef int      i4   __attribute__((ext_vector_type(4)));
typedef float    f4   __attribute__((ext_vector_type(4)));
typedef _Float16 h4   __attribute__((ext_vector_type(4)));
typedef _Float16 v8h  __attribute__((ext_vector_type(8)));
typedef _Float16 v16h __attribute__((ext_vector_type(16)));
typedef float    v8f  __attribute__((ext_vector_type(8)));
typedef int      gi4v __attribute__((vector_size(16)));   // matches async-LDS builtin param

#define D_DIM 256
#define LDS_PITCH 264                       // 256 + 8 halfs pad: 528B = 132 dwords (≡4 mod 64 banks)
#define LDS_BYTES (D_DIM * LDS_PITCH * 2)   // 135168 B

#define AS1 __attribute__((address_space(1)))
#define AS3 __attribute__((address_space(3)))

// ---------- fp32 -> fp16 conversion (vectorized by 4) ----------
__global__ void cvt_f16_kernel(const f4* __restrict__ x, h4* __restrict__ out, int nvec) {
    int i = blockIdx.x * blockDim.x + threadIdx.x;
    if (i < nvec) {
        f4 f = x[i];
        h4 h;
        h.x = (_Float16)f.x; h.y = (_Float16)f.y;
        h.z = (_Float16)f.z; h.w = (_Float16)f.w;
        out[i] = h;
    }
}

// ---------- row sums of Wn [256 x 256]: out[m] = sum_k Wn[m,k] ----------
__global__ void rowsum_kernel(const float* __restrict__ w, float* __restrict__ out) {
    int m = threadIdx.x;  // one block of 256
    const f4* r = (const f4*)(w + (size_t)m * D_DIM);
    float s = 0.f;
    #pragma unroll 4
    for (int k = 0; k < D_DIM / 4; k++) {
        f4 f = r[k];
        s += f.x + f.y + f.z + f.w;
    }
    out[m] = s;
}

// ---------- extract column 0 of x [N x 256] ----------
__global__ void extract_col0(const float* __restrict__ x, float* __restrict__ col, int n) {
    int j = blockIdx.x * blockDim.x + threadIdx.x;
    if (j < n) col[j] = x[(size_t)j * D_DIM];
}

// ---------- adjacency row reduction: MODE 0 = masked max, MODE 1 = masked sum ----------
// One block (8 waves) per adjacency row; streaming int4 loads (NT) + prefetch.
template<int MODE>
__global__ __launch_bounds__(256) void neigh_kernel(const i4* __restrict__ adj,
                                                    const f4* __restrict__ col,
                                                    float* __restrict__ out, int nvec) {
    size_t row = blockIdx.x;
    const i4* arow = adj + row * (size_t)nvec;
    float m = -FLT_MAX, s = 0.f;
    int any = 0;
    for (int i = threadIdx.x; i < nvec; i += blockDim.x) {
        i4 a = __builtin_nontemporal_load(&arow[i]);
        f4 c = col[i];
        __builtin_prefetch(&arow[i + blockDim.x], 0, 0);  // global_prefetch_b8
        if (MODE == 0) {
            if (a.x > 0) { any = 1; m = fmaxf(m, c.x); }
            if (a.y > 0) { any = 1; m = fmaxf(m, c.y); }
            if (a.z > 0) { any = 1; m = fmaxf(m, c.z); }
            if (a.w > 0) { any = 1; m = fmaxf(m, c.w); }
        } else {
            any |= (a.x | a.y | a.z | a.w);
            s += (float)a.x * c.x + (float)a.y * c.y + (float)a.z * c.z + (float)a.w * c.w;
        }
    }
    // wave32 reduction
    #pragma unroll
    for (int off = 16; off >= 1; off >>= 1) {
        if (MODE == 0) m = fmaxf(m, __shfl_xor(m, off, 32));
        else           s += __shfl_xor(s, off, 32);
        any |= __shfl_xor(any, off, 32);
    }
    __shared__ float red[8];
    __shared__ int   ra[8];
    int wv = threadIdx.x >> 5;
    if ((threadIdx.x & 31) == 0) { red[wv] = (MODE == 0) ? m : s; ra[wv] = any; }
    __syncthreads();
    if (threadIdx.x == 0) {
        float R = red[0];
        int   A = ra[0];
        #pragma unroll
        for (int w = 1; w < 8; w++) {
            R = (MODE == 0) ? fmaxf(R, red[w]) : (R + red[w]);
            A |= ra[w];
        }
        out[row] = A ? R : 0.f;
    }
}

// ---------- fused GEMM: Y = act(Xh @ Wh^T + neigh[:,None] * wsum[None,:]) ----------
// Xh: [N x 256] f16 row-major.  Wh: [256 x 256] f16 row-major (Y[n,m] = sum_k X[n,k]*W[m,k]).
// The whole W (f16) is staged in LDS once per block via async global->LDS loads
// (all 8 waves share it); each wave owns a 16-row tile, holds all 8 A K-fragments
// in VGPRs, and sweeps the 16 column tiles reading B from LDS (padded pitch).
template<int RELU>
__global__ __launch_bounds__(256) void gemm_wmma_kernel(const _Float16* __restrict__ Xh,
                                                        const _Float16* __restrict__ Wh,
                                                        const float* __restrict__ neigh,
                                                        const float* __restrict__ wsum,
                                                        float* __restrict__ Y,
                                                        int ntiles) {
    extern __shared__ _Float16 Blds[];  // [256][LDS_PITCH], 132KB dynamic LDS

    // ---- cooperative stage of W into LDS (16B chunks; 8192 chunks / 256 threads) ----
    {
        const int nchunks = (D_DIM * D_DIM) / 8;  // 8 halfs = 16B per chunk
        for (int i = threadIdx.x; i < nchunks; i += 256) {
            int row  = i >> 5;             // 32 chunks per 256-half row
            int coli = (i & 31) << 3;      // half offset within row
            const _Float16* gp = Wh   + (size_t)row * D_DIM     + coli;
            _Float16*       lp = Blds + (size_t)row * LDS_PITCH + coli;
#if __has_builtin(__builtin_amdgcn_global_load_async_to_lds_b128)
            __builtin_amdgcn_global_load_async_to_lds_b128(
                (AS1 gi4v*)gp, (AS3 gi4v*)lp, 0, 0);
#else
            *(v8h*)lp = *(const v8h*)gp;
#endif
        }
#if __has_builtin(__builtin_amdgcn_global_load_async_to_lds_b128)
#if __has_builtin(__builtin_amdgcn_s_wait_asynccnt)
        __builtin_amdgcn_s_wait_asynccnt(0);
#else
        asm volatile("s_wait_asynccnt 0x0" ::: "memory");
#endif
#endif
        __syncthreads();
    }

    int wv   = threadIdx.x >> 5;
    int lane = threadIdx.x & 31;
    int t = blockIdx.x * 8 + wv;
    if (t >= ntiles) return;          // uniform per wave -> EXEC all-1 for WMMA waves
    int r0 = t * 16;
    int mrow = lane & 15;             // M (A) / N (B,C) index within tile
    int hi   = lane >> 4;             // lane-half selector

    // A fragments: lane holds row r0+mrow; element h -> K = hi*16 + h
    union { v16h v; v8h h[2]; } a[8];
    const _Float16* arow = Xh + (size_t)(r0 + mrow) * D_DIM;
    #pragma unroll
    for (int ks = 0; ks < 8; ks++) {
        int k0 = ks * 32;
        a[ks].h[0] = *(const v8h*)(arow + k0 + hi * 16);
        a[ks].h[1] = *(const v8h*)(arow + k0 + hi * 16 + 8);
    }

    // neigh values for the 8 output rows this lane owns (C layout: M = v + hi*8)
    float ng[8];
    #pragma unroll
    for (int v = 0; v < 8; v++) ng[v] = neigh[r0 + v + hi * 8];

    for (int ct = 0; ct < 16; ct++) {
        int c0 = ct * 16;
        // B fragment from LDS: N = lane%16 -> W row c0+mrow; element h -> K = k0 + hi*16 + h
        const _Float16* wrow = Blds + (size_t)(c0 + mrow) * LDS_PITCH + hi * 16;
        v8f acc = {0.f, 0.f, 0.f, 0.f, 0.f, 0.f, 0.f, 0.f};
        #pragma unroll
        for (int ks = 0; ks < 8; ks++) {
            v16h b = *(const v16h*)(wrow + ks * 32);
            acc = __builtin_amdgcn_wmma_f32_16x16x32_f16(
                false, a[ks].v, false, b, (short)0, acc, false, false);
        }
        int col = c0 + mrow;
        float ws = wsum[col];
        #pragma unroll
        for (int v = 0; v < 8; v++) {
            int row = r0 + v + hi * 8;
            float val = acc[v] + ng[v] * ws;
            if (RELU) val = fmaxf(val, 0.f);
            Y[(size_t)row * D_DIM + col] = val;
        }
    }
}

// ---------------------------------------------------------------------------
extern "C" void kernel_launch(void* const* d_in, const int* in_sizes, int n_in,
                              void* d_out, int out_size, void* d_ws, size_t ws_size,
                              hipStream_t stream) {
    const float* x1   = (const float*)d_in[0];
    const float* x2   = (const float*)d_in[1];
    const int* adj12  = (const int*)d_in[2];   // adj_1to2 [N,N]
    const int* adj21  = (const int*)d_in[3];   // adj_2to1 [N,N]
    const float* Wmat[8] = {
        (const float*)d_in[4],  (const float*)d_in[5],   // w1s0, w1n0
        (const float*)d_in[6],  (const float*)d_in[7],   // w2s0, w2n0
        (const float*)d_in[8],  (const float*)d_in[9],   // w1s1, w1n1
        (const float*)d_in[10], (const float*)d_in[11]}; // w2s1, w2n1

    const int N  = in_sizes[0] / D_DIM;        // 10000
    const size_t ND = (size_t)N * D_DIM;
    const int nvec   = N / 4;                  // int4/float4 per adj row
    const int ntiles = N / 16;                 // 625
    const int gemm_blocks = (ntiles + 7) / 8;

    // ---- carve workspace (256B-aligned chunks) ----
    size_t off = 0;
    auto carve = [&](size_t bytes) -> void* {
        void* p = (char*)d_ws + off;
        off += (bytes + 255) & ~(size_t)255;
        return p;
    };
    _Float16* xh1 = (_Float16*)carve(ND * 2);
    _Float16* xh2 = (_Float16*)carve(ND * 2);
    _Float16* wh[4];
    for (int i = 0; i < 4; i++) wh[i] = (_Float16*)carve((size_t)D_DIM * D_DIM * 2);
    float* col1 = (float*)carve((size_t)N * 4);
    float* col2 = (float*)carve((size_t)N * 4);
    float* ne1  = (float*)carve((size_t)N * 4);
    float* ne2  = (float*)carve((size_t)N * 4);
    float* wsum[4];
    for (int i = 0; i < 4; i++) wsum[i] = (float*)carve(D_DIM * 4);
    (void)ws_size; (void)n_in; (void)out_size;

    float* out1 = (float*)d_out;
    float* out2 = out1 + ND;

    // ---- one-time weight prep: Ws -> f16, rowsum(Wn) ----
    const int wvec = D_DIM * D_DIM / 4;
    const int ws_idx[4] = {0, 2, 4, 6};  // w1s0, w2s0, w1s1, w2s1
    const int wn_idx[4] = {1, 3, 5, 7};  // w1n0, w2n0, w1n1, w2n1
    for (int i = 0; i < 4; i++) {
        cvt_f16_kernel<<<(wvec + 255) / 256, 256, 0, stream>>>(
            (const f4*)Wmat[ws_idx[i]], (h4*)wh[i], wvec);
        rowsum_kernel<<<1, 256, 0, stream>>>(Wmat[wn_idx[i]], wsum[i]);
    }

    const int xvec = (int)(ND / 4);
    for (int layer = 0; layer < 2; layer++) {
        const float* X1 = (layer == 0) ? x1 : out1;
        const float* X2 = (layer == 0) ? x2 : out2;

        // node scalars
        extract_col0<<<(N + 255) / 256, 256, 0, stream>>>(X1, col1, N);
        extract_col0<<<(N + 255) / 256, 256, 0, stream>>>(X2, col2, N);
        // neigh1[i] = masked max over adj_2to1 row of x2[:,0]
        neigh_kernel<0><<<N, 256, 0, stream>>>((const i4*)adj21, (const f4*)col2, ne1, nvec);
        // neigh2[j] = masked sum adj_1to2 @ x1[:,0]
        neigh_kernel<1><<<N, 256, 0, stream>>>((const i4*)adj12, (const f4*)col1, ne2, nvec);

        // f16 copies of activations
        cvt_f16_kernel<<<(xvec + 255) / 256, 256, 0, stream>>>((const f4*)X1, (h4*)xh1, xvec);
        cvt_f16_kernel<<<(xvec + 255) / 256, 256, 0, stream>>>((const f4*)X2, (h4*)xh2, xvec);

        // fused WMMA GEMMs (layer 0: ReLU, layer 1: identity), in-place into d_out
        int w0 = (layer == 0) ? 0 : 2;  // wh/wsum index for side 1
        int w1 = (layer == 0) ? 1 : 3;  // wh/wsum index for side 2
        if (layer == 0) {
            gemm_wmma_kernel<1><<<gemm_blocks, 256, LDS_BYTES, stream>>>(xh1, wh[w0], ne1, wsum[w0], out1, ntiles);
            gemm_wmma_kernel<1><<<gemm_blocks, 256, LDS_BYTES, stream>>>(xh2, wh[w1], ne2, wsum[w1], out2, ntiles);
        } else {
            gemm_wmma_kernel<0><<<gemm_blocks, 256, LDS_BYTES, stream>>>(xh1, wh[w0], ne1, wsum[w0], out1, ntiles);
            gemm_wmma_kernel<0><<<gemm_blocks, 256, LDS_BYTES, stream>>>(xh2, wh[w1], ne2, wsum[w1], out2, ntiles);
        }
    }
}
